// TFAM_4853313044766
// MI455X (gfx1250) — matrix operand order repaired and checked by reading, not compiled, not run on hardware
//
#include <hip/hip_runtime.h>
#include <hip/hip_bf16.h>

// TFAM on MI455X (gfx1250): all contractions via v_wmma_f32_16x16x32_bf16.
// B=8, C=256, H=W=64, HW=4096, K=64.
//   pack   : f32 -> bf16 copies (vectorized) + fused bias array
//   proj   : [128,256]x[256,4096] per batch -> top||center (bf16)
//   conv3x3: halo-tile implicit GEMM, 3 dy stages x 3 dx taps sharing one B tile
//   scores : phase0 = global max via atomicMax; phase1 = P=exp(s-M) bf16 + Z atomicAdd
//   pv     : O = (P x V)/Z, f32 [HW,C]  (raw reshape => flat elementwise add with x)
// A-tile staging uses GLOBAL_LOAD_ASYNC_TO_LDS_B128 (ASYNCcnt) when the builtin
// exists; the B-tile register-transpose overlaps the async copy.
// Workspace ~365 MB (dominated by P: 8*4096*4096*2 = 268 MB).

#define Bn 8
#define Cn 256
#define Hn 64
#define Wn 64
#define HWn 4096

typedef __attribute__((ext_vector_type(16))) __bf16 v16bf;
typedef __attribute__((ext_vector_type(8)))  float  v8f;
typedef unsigned short ushort_t;

#if defined(__has_builtin)
#if __has_builtin(__builtin_amdgcn_global_load_async_to_lds_b128) && \
    __has_builtin(__builtin_amdgcn_s_wait_asynccnt)
#define HAVE_ASYNC_LDS 1
#endif
#endif
#ifndef HAVE_ASYNC_LDS
#define HAVE_ASYNC_LDS 0
#endif

#if HAVE_ASYNC_LDS
#define GLOBAL_AS __attribute__((address_space(1)))
#define LDS_AS    __attribute__((address_space(3)))
typedef int v4i_t __attribute__((vector_size(16)));
static __device__ __forceinline__ void copy16ToLds(const __bf16* g, __bf16* l) {
    __builtin_amdgcn_global_load_async_to_lds_b128(
        (GLOBAL_AS v4i_t*)g, (LDS_AS v4i_t*)l, 0, 0);
}
static __device__ __forceinline__ void asyncWait() { __builtin_amdgcn_s_wait_asynccnt(0); }
#else
static __device__ __forceinline__ void copy16ToLds(const __bf16* g, __bf16* l) {
    *(uint4*)l = *(const uint4*)g;
}
static __device__ __forceinline__ void asyncWait() {}
#endif

static __device__ __forceinline__ __bf16 f2bf(float f) {
    unsigned u = __builtin_bit_cast(unsigned, f);
    unsigned r = 0x7FFFu + ((u >> 16) & 1u);           // round-to-nearest-even
    unsigned short h = (unsigned short)((u + r) >> 16);
    return __builtin_bit_cast(__bf16, h);
}
static __device__ __forceinline__ unsigned short f2bfu(float f) {
    return __builtin_bit_cast(unsigned short, f2bf(f));
}

// ---- WMMA fragment helpers (layouts per CDNA5 ISA 7.12.2, 16-bit, wave32) ----
static __device__ __forceinline__ int kofA(int e, int half) {
    int r = e >> 1, j = e & 1;
    return ((r & 4) ? 16 : 0) + half * 8 + (r & 3) * 2 + j;
}
// A tile row-major [row][32]: frag = 32 contiguous bytes -> 2x ds_load_b128
static __device__ __forceinline__ v16bf loadFragA(const __bf16* lds, int row, int lane) {
    int half = lane >> 4;
    v16bf a;
#pragma unroll
    for (int e = 0; e < 16; ++e) a[e] = lds[row * 32 + half * 16 + e];
    return a;
}
// B tile column-major [col][32]: frag = 32 contiguous bytes -> 2x ds_load_b128
static __device__ __forceinline__ v16bf loadFragBT(const __bf16* lds, int col, int lane) {
    const __bf16* p = &lds[col * 32 + (lane >> 4) * 16];
    v16bf b;
#pragma unroll
    for (int e = 0; e < 16; ++e) b[e] = p[e];
    return b;
}
static __device__ __forceinline__ v8f wmma_bf16(v16bf a, v16bf b, v8f c) {
    return __builtin_amdgcn_wmma_f32_16x16x32_bf16(false, a, false, b, (short)0, c, false, false);
}

// stage A tile: 64 rows x 32 k, global row-major (row stride gs elems) -> LDS [64][32]
// (async global->LDS when available; no VGPR round trip)
static __device__ __forceinline__ void stageA(const __bf16* g, int gs, __bf16* lds, int tid) {
    int r = tid >> 1, ko = (tid & 1) * 16;
    copy16ToLds(g + (size_t)r * gs + ko,     &lds[r * 32 + ko]);
    copy16ToLds(g + (size_t)r * gs + ko + 8, &lds[r * 32 + ko + 8]);
}
// stage+transpose B tile: 32 rows(k) x 64 cols, row-major (stride gs) -> LDS [64 cols][32 k]
static __device__ __forceinline__ void stageBT(const __bf16* g, int gs, __bf16* lds, int tid) {
    int rp = tid >> 3;            // 0..15 (k pair)
    int cb = (tid & 7) * 8;       // col base
    ushort_t a0[8], a1[8];
    *(uint4*)a0 = *(const uint4*)(g + (size_t)(2 * rp) * gs + cb);
    *(uint4*)a1 = *(const uint4*)(g + (size_t)(2 * rp + 1) * gs + cb);
#pragma unroll
    for (int c = 0; c < 8; ++c) {
        unsigned v = (unsigned)a0[c] | ((unsigned)a1[c] << 16);
        *(unsigned*)(&lds[(cb + c) * 32 + 2 * rp]) = v;
    }
}

static __device__ __forceinline__ void atomicMaxF(float* addr, float v) {
    if (v >= 0.0f) atomicMax((int*)addr, __float_as_int(v));
    else           atomicMin((unsigned int*)addr, __float_as_uint(v));
}

// -------------------------------- pack --------------------------------------
__global__ void pack_kernel(const float* __restrict__ x,
                            const float* __restrict__ topw, const float* __restrict__ cenw,
                            const float* __restrict__ botw, const float* __restrict__ outw,
                            const float* __restrict__ topb, const float* __restrict__ cenb,
                            __bf16* __restrict__ xb, __bf16* __restrict__ wtc,
                            __bf16* __restrict__ wcb, __bf16* __restrict__ wco,
                            float* __restrict__ biastc, float* __restrict__ stats) {
    unsigned tid = blockIdx.x * blockDim.x + threadIdx.x;
    unsigned nt  = gridDim.x * blockDim.x;
    const float4* x4 = (const float4*)x;
    for (unsigned i = tid; i < (unsigned)(Bn * Cn * HWn) / 8u; i += nt) {
        float4 a = x4[2 * i], c = x4[2 * i + 1];
        ushort_t pk[8] = { f2bfu(a.x), f2bfu(a.y), f2bfu(a.z), f2bfu(a.w),
                           f2bfu(c.x), f2bfu(c.y), f2bfu(c.z), f2bfu(c.w) };
        ((uint4*)xb)[i] = *(uint4*)pk;
    }
    for (unsigned i = tid; i < 128u * 256u; i += nt) {
        int m = i >> 8, k = i & 255;
        wtc[i] = f2bf(m < 64 ? topw[m * 256 + k] : cenw[(m - 64) * 256 + k]);
    }
    for (unsigned i = tid; i < 9u * 256u * 256u; i += nt) {           // [(t*256+o)*256+ic]
        int t = i >> 16, o = (i >> 8) & 255, ic = i & 255;
        wcb[i] = f2bf(botw[(o * 256 + ic) * 9 + t]);
        wco[i] = f2bf(outw[(o * 256 + ic) * 9 + t]);
    }
    if (tid < 128) biastc[tid] = (tid < 64) ? topb[tid] : cenb[tid - 64];
    if (tid < Bn) { stats[2 * tid] = -__builtin_inff(); stats[2 * tid + 1] = 0.0f; }
}

// ------------------------- proj: wtc x x -> top||center ---------------------
__global__ __launch_bounds__(128) void proj_kernel(const __bf16* __restrict__ xb,
                                                   const __bf16* __restrict__ wtc,
                                                   const float* __restrict__ biastc,
                                                   __bf16* __restrict__ proj) {
    __shared__ __bf16 sA[64 * 32];   // [row][k]
    __shared__ __bf16 sB[64 * 32];   // [col][k] (transposed)
    int nt = blockIdx.x, mt = blockIdx.y, b = blockIdx.z;
    int tid = threadIdx.x, lane = tid & 31, wv = tid >> 5;
    const __bf16* Bsrc = xb + (size_t)b * Cn * HWn + nt * 64;
    v8f acc[4] = {};
    for (int k0 = 0; k0 < 256; k0 += 32) {
        stageA(wtc + (size_t)(mt * 64) * 256 + k0, 256, sA, tid);   // async -> LDS
        stageBT(Bsrc + (size_t)k0 * HWn, HWn, sB, tid);             // overlaps async
        asyncWait();
        __syncthreads();
        v16bf af = loadFragA(sA, wv * 16 + (lane & 15), lane);
#pragma unroll
        for (int nb = 0; nb < 4; ++nb)
            acc[nb] = wmma_bf16(af, loadFragBT(sB, nb * 16 + (lane & 15), lane), acc[nb]);
        __syncthreads();
    }
    int rowbase = mt * 64 + wv * 16 + ((lane >> 4) << 3);
    float bs[8];
#pragma unroll
    for (int r = 0; r < 8; ++r) bs[r] = biastc[rowbase + r];
#pragma unroll
    for (int nb = 0; nb < 4; ++nb)
#pragma unroll
        for (int r = 0; r < 8; ++r) {
            int col = nt * 64 + nb * 16 + (lane & 15);
            proj[((size_t)b * 128 + rowbase + r) * HWn + col] = f2bf(acc[nb][r] + bs[r]);
        }
}

// -------------- conv3x3: halo-tile implicit GEMM (3 dy x 3 dx) --------------
// mode 0: write bf16 V[b][p][o] (transposed, for P@V);  mode 1: write f32 NCHW
__global__ __launch_bounds__(128) void conv3x3_kernel(const __bf16* __restrict__ in,
                                                      const __bf16* __restrict__ wt,
                                                      const float* __restrict__ bias,
                                                      void* __restrict__ outp, int mode) {
    __shared__ __bf16 sA[3][64 * 32];    // one A tile per dx tap
    __shared__ __bf16 sB[66 * 32];       // [col 0..65][k], halo cols 0 and 65
    int y = blockIdx.x, mt = blockIdx.y, b = blockIdx.z;
    int tid = threadIdx.x, lane = tid & 31, wv = tid >> 5;
    v8f acc[4] = {};
    for (int dy = -1; dy <= 1; ++dy) {
        int sy = y + dy;
        if (sy < 0 || sy >= Hn) continue;                 // uniform per block
        for (int k0 = 0; k0 < 256; k0 += 32) {
#pragma unroll
            for (int dx = 0; dx < 3; ++dx) {
                int t = (dy + 1) * 3 + dx;
                stageA(wt + ((size_t)(t * 256 + mt * 64)) * 256 + k0, 256, sA[dx], tid);
            }
            stageBT(in + ((size_t)b * Cn + k0) * HWn + sy * Wn, HWn, sB + 1 * 32, tid);
            if (tid < 32) {                               // zero halo columns
                int col = (tid < 16) ? 0 : 65;
                *(unsigned*)&sB[col * 32 + 2 * (tid & 15)] = 0u;
            }
            asyncWait();
            __syncthreads();
#pragma unroll
            for (int dx = 0; dx < 3; ++dx) {
                v16bf af = loadFragA(sA[dx], wv * 16 + (lane & 15), lane);
#pragma unroll
                for (int nb = 0; nb < 4; ++nb) {
                    int col = nb * 16 + (lane & 15) + dx;  // x + (dx-1) + 1 (halo offset)
                    acc[nb] = wmma_bf16(af, loadFragBT(sB, col, lane), acc[nb]);
                }
            }
            __syncthreads();
        }
    }
    int o0 = mt * 64 + wv * 16 + ((lane >> 4) << 3);      // 8 contiguous output channels
    float bs[8];
#pragma unroll
    for (int r = 0; r < 8; ++r) bs[r] = bias[o0 + r];
    if (mode == 0) {
#pragma unroll
        for (int nb = 0; nb < 4; ++nb) {
            int p = y * Wn + nb * 16 + (lane & 15);
            ushort_t pk[8];
#pragma unroll
            for (int r = 0; r < 8; ++r) pk[r] = f2bfu(acc[nb][r] + bs[r]);
            *(uint4*)((__bf16*)outp + ((size_t)b * HWn + p) * Cn + o0) = *(uint4*)pk;
        }
    } else {
#pragma unroll
        for (int nb = 0; nb < 4; ++nb)
#pragma unroll
            for (int r = 0; r < 8; ++r) {
                int p = y * Wn + nb * 16 + (lane & 15);
                ((float*)outp)[((size_t)b * Cn + o0 + r) * HWn + p] = acc[nb][r] + bs[r];
            }
    }
}

// ------------- scores: S = center^T x top; phase0=max, phase1=exp -----------
__global__ __launch_bounds__(128) void attn_scores_kernel(const __bf16* __restrict__ proj,
                                                          float* __restrict__ stats,
                                                          __bf16* __restrict__ Pout, int phase) {
    __shared__ __bf16 sA[64 * 32];   // [n][k]  (center, transposed on store)
    __shared__ __bf16 sB[64 * 32];   // [m][k]  (top, transposed on store)
    int mtile = blockIdx.x, ntile = blockIdx.y, b = blockIdx.z;
    int tid = threadIdx.x, lane = tid & 31, wv = tid >> 5;
    const __bf16* top = proj + (size_t)b * 128 * HWn;
    const __bf16* cen = top + (size_t)64 * HWn;
    v8f acc[4] = {};
    for (int k0 = 0; k0 < 64; k0 += 32) {
        stageBT(cen + (size_t)k0 * HWn + ntile * 64, HWn, sA, tid);  // -> [n][k] == A row-major
        stageBT(top + (size_t)k0 * HWn + mtile * 64, HWn, sB, tid);
        __syncthreads();
        v16bf af = loadFragA(sA, wv * 16 + (lane & 15), lane);
#pragma unroll
        for (int nb = 0; nb < 4; ++nb)
            acc[nb] = wmma_bf16(af, loadFragBT(sB, nb * 16 + (lane & 15), lane), acc[nb]);
        __syncthreads();
    }
    if (phase == 0) {
        float m = -__builtin_inff();
#pragma unroll
        for (int nb = 0; nb < 4; ++nb)
#pragma unroll
            for (int r = 0; r < 8; ++r) m = fmaxf(m, acc[nb][r]);
        for (int off = 16; off; off >>= 1) m = fmaxf(m, __shfl_xor(m, off, 32));
        if (lane == 0) atomicMaxF(&stats[2 * b], m);
    } else {
        float Mb = stats[2 * b];
        float zs = 0.0f;
        int nbase = ntile * 64 + wv * 16 + ((lane >> 4) << 3);
#pragma unroll
        for (int nb = 0; nb < 4; ++nb) {
            int m = mtile * 64 + nb * 16 + (lane & 15);
#pragma unroll
            for (int r = 0; r < 8; ++r) {
                float e = __expf(acc[nb][r] - Mb);
                zs += e;
                Pout[((size_t)b * HWn + nbase + r) * HWn + m] = f2bf(e);
            }
        }
        for (int off = 16; off; off >>= 1) zs += __shfl_xor(zs, off, 32);
        if (lane == 0) atomicAdd(&stats[2 * b + 1], zs);
    }
}

// --------------------------- O = (P x V) / Z --------------------------------
__global__ __launch_bounds__(128) void pv_kernel(const __bf16* __restrict__ P,
                                                 const __bf16* __restrict__ V,
                                                 const float* __restrict__ stats,
                                                 float* __restrict__ O) {
    __shared__ __bf16 sA[64 * 32];
    __shared__ __bf16 sB[64 * 32];
    int ct = blockIdx.x, nt = blockIdx.y, b = blockIdx.z;
    int tid = threadIdx.x, lane = tid & 31, wv = tid >> 5;
    float inv = 1.0f / stats[2 * b + 1];
    const __bf16* Pb = P + (size_t)b * HWn * HWn + (size_t)(nt * 64) * HWn;
    const __bf16* Vb = V + (size_t)b * HWn * Cn + ct * 64;
    v8f acc[4] = {};
    for (int k0 = 0; k0 < HWn; k0 += 32) {
        __builtin_prefetch((const void*)(Pb + (size_t)(tid >> 1) * HWn + k0 + 64), 0, 0);
        stageA(Pb + k0, HWn, sA, tid);                 // async -> LDS (268 MB stream)
        stageBT(Vb + (size_t)k0 * Cn, Cn, sB, tid);    // overlaps async
        asyncWait();
        __syncthreads();
        v16bf af = loadFragA(sA, wv * 16 + (lane & 15), lane);
#pragma unroll
        for (int nb = 0; nb < 4; ++nb)
            acc[nb] = wmma_bf16(af, loadFragBT(sB, nb * 16 + (lane & 15), lane), acc[nb]);
        __syncthreads();
    }
    int nbase = nt * 64 + wv * 16 + ((lane >> 4) << 3);
#pragma unroll
    for (int nb = 0; nb < 4; ++nb) {
        int c = ct * 64 + nb * 16 + (lane & 15);
#pragma unroll
        for (int r = 0; r < 8; ++r)
            O[((size_t)b * HWn + nbase + r) * Cn + c] = acc[nb][r] * inv;
    }
}

// --------------------- y = bf16(x + O_flat) (raw reshape) -------------------
__global__ void addpack_kernel(const float* __restrict__ x, const float* __restrict__ O,
                               __bf16* __restrict__ y) {
    unsigned tid = blockIdx.x * blockDim.x + threadIdx.x;
    unsigned nt  = gridDim.x * blockDim.x;
    const float4* x4 = (const float4*)x;
    const float4* O4 = (const float4*)O;
    for (unsigned i = tid; i < (unsigned)(Bn * Cn * HWn) / 4u; i += nt) {
        float4 a = x4[i], o = O4[i];
        ushort_t pk[4] = { f2bfu(a.x + o.x), f2bfu(a.y + o.y), f2bfu(a.z + o.z), f2bfu(a.w + o.w) };
        ((uint2*)y)[i] = *(uint2*)pk;
    }
}

// ------------------------------- launch -------------------------------------
extern "C" void kernel_launch(void* const* d_in, const int* in_sizes, int n_in,
                              void* d_out, int out_size, void* d_ws, size_t ws_size,
                              hipStream_t stream) {
    const float* x    = (const float*)d_in[0];
    const float* topw = (const float*)d_in[1];
    const float* topb = (const float*)d_in[2];
    const float* cenw = (const float*)d_in[3];
    const float* cenb = (const float*)d_in[4];
    const float* botw = (const float*)d_in[5];
    const float* botb = (const float*)d_in[6];
    const float* outw = (const float*)d_in[7];
    const float* outb = (const float*)d_in[8];

    char* w = (char*)d_ws;
    size_t off = 0;
    auto take = [&](size_t bytes) { void* p = w + off; off = (off + bytes + 255) & ~(size_t)255; return p; };
    __bf16* xb    = (__bf16*)take((size_t)Bn * Cn * HWn * 2);   // 16.8 MB
    __bf16* ybuf  = (__bf16*)take((size_t)Bn * Cn * HWn * 2);   // 16.8 MB
    __bf16* proj  = (__bf16*)take((size_t)Bn * 128 * HWn * 2);  //  8.4 MB
    __bf16* vbot  = (__bf16*)take((size_t)Bn * HWn * Cn * 2);   // 16.8 MB
    float*  Obuf  = (float*) take((size_t)Bn * HWn * Cn * 4);   // 33.6 MB
    __bf16* wtc   = (__bf16*)take((size_t)128 * 256 * 2);
    __bf16* wcb   = (__bf16*)take((size_t)9 * 256 * 256 * 2);
    __bf16* wco   = (__bf16*)take((size_t)9 * 256 * 256 * 2);
    float*  biastc= (float*) take((size_t)128 * 4);
    float*  stats = (float*) take((size_t)Bn * 2 * 4);
    __bf16* Pm    = (__bf16*)take((size_t)Bn * HWn * HWn * 2);  // 268 MB
    (void)ws_size; (void)in_sizes; (void)n_in; (void)out_size;

    pack_kernel<<<2048, 256, 0, stream>>>(x, topw, cenw, botw, outw, topb, cenb,
                                          xb, wtc, wcb, wco, biastc, stats);
    proj_kernel<<<dim3(64, 2, Bn), 128, 0, stream>>>(xb, wtc, biastc, proj);
    conv3x3_kernel<<<dim3(64, 4, Bn), 128, 0, stream>>>(xb, wcb, botb, (void*)vbot, 0);
    attn_scores_kernel<<<dim3(64, 64, Bn), 128, 0, stream>>>(proj, stats, Pm, 0);
    attn_scores_kernel<<<dim3(64, 64, Bn), 128, 0, stream>>>(proj, stats, Pm, 1);
    pv_kernel<<<dim3(4, 64, Bn), 128, 0, stream>>>(Pm, vbot, stats, Obuf);
    addpack_kernel<<<4096, 256, 0, stream>>>(x, Obuf, ybuf);
    conv3x3_kernel<<<dim3(64, 4, Bn), 128, 0, stream>>>(ybuf, wco, outb, d_out, 1);
}
